// EnhancedGNNModel_51582557225194
// MI455X (gfx1250) — matrix operand order repaired and checked by reading, not compiled, not run on hardware
//
#include <hip/hip_runtime.h>

#define N_NODES 50000
#define N_EDGES 800000
#define DIM     128
#define N_GRAPHS 8

typedef _Float16 v16h __attribute__((ext_vector_type(16)));
typedef _Float16 h8   __attribute__((ext_vector_type(8)));
typedef float    v8f  __attribute__((ext_vector_type(8)));

// ---------------------------------------------------------------------------
// Transposed formulation: D = A x B with A = W^T (static, pre-swizzled into
// the CDNA5 16x32 f16 A-fragment layout) and B = activations^T, whose
// B-fragment for lane (half=lane>>4, c=lane&15) is two runs of 8 halfs
// of row c of the row-major activation table:  row + kt*32 + 8*half (+16).
// D element (i, lane) = output[feat = ft*16 + i + 8*half][node/edge = c],
// so each lane owns 8 consecutive features of one node/edge -> packed stores.
// ---------------------------------------------------------------------------
__device__ __forceinline__ v16h mk16(h8 lo, h8 hi) {
  union { v16h v; h8 h[2]; } u;
  u.h[0] = lo; u.h[1] = hi;
  return u.v;
}
// B-fragment from a row-major f16 row pointer (global or LDS)
__device__ __forceinline__ v16h load_b_row(const _Float16* row, int kt, int half) {
  const h8* p = reinterpret_cast<const h8*>(row + kt*32 + 8*half);
  return mk16(p[0], p[2]);
}
// A-fragment (weights) from pre-swizzled storage: 16 contiguous halfs/lane
__device__ __forceinline__ v16h load_w_frag(const _Float16* sW, int kt, int ft, int lane) {
  const h8* p = reinterpret_cast<const h8*>(sW + ((ft*4 + kt) << 9) + (lane << 4));
  return mk16(p[0], p[1]);
}
__device__ __forceinline__ v8f wmma16(v16h a, v16h b, v8f c) {
  return __builtin_amdgcn_wmma_f32_16x16x32_f16(false, a, false, b, (short)0, c, false, false);
}
__device__ __forceinline__ void atomic_add_f32(float* p, float v) {
  (void)__hip_atomic_fetch_add(p, v, __ATOMIC_RELAXED, __HIP_MEMORY_SCOPE_AGENT);
}
// init accumulator from 8 consecutive f32 (bias vector slice)
__device__ __forceinline__ v8f load_bias8(const float* b) {
  const float4 lo = *reinterpret_cast<const float4*>(b);
  const float4 hi = *reinterpret_cast<const float4*>(b + 4);
  v8f c;
  c[0]=lo.x; c[1]=lo.y; c[2]=lo.z; c[3]=lo.w;
  c[4]=hi.x; c[5]=hi.y; c[6]=hi.z; c[7]=hi.w;
  return c;
}
__device__ __forceinline__ h8 relu_pack(v8f c) {
  h8 v;
#pragma unroll
  for (int i = 0; i < 8; ++i) v[i] = (_Float16)fmaxf(c[i], 0.0f);
  return v;
}

// ---------------------------------------------------------------------------
// Weight swizzle: 15 matrices 128x128 f32 -> f16 A-fragment layout of W^T.
// A row = output feature n, A K-dim = input k.  element j of v16h:
//   k = kt*32 + 8*(lane>>4) + j + (j & 8)
// m: 0-2 nW1[l], 3-5 nW2[l], 6-8 eW1[l] rows 0..127 (dst), 9-11 eW1[l] rows
// 128..255 (src), 12-14 eW2[l].
// ---------------------------------------------------------------------------
__global__ void prep_weights_kernel(const float* __restrict__ nW1,
                                    const float* __restrict__ nW2,
                                    const float* __restrict__ eW1,
                                    const float* __restrict__ eW2,
                                    _Float16* __restrict__ w16) {
  int t = blockIdx.x * 256 + threadIdx.x;
  if (t >= 15 * 16384) return;
  int m = t / 16384, e = t % 16384;
  int tile = e >> 9;
  int lane = (e >> 4) & 31;
  int j = e & 15;
  int kt = tile & 3, ft = tile >> 2;
  int k = kt * 32 + ((lane >> 4) << 3) + j + (j & 8);
  int n = ft * 16 + (lane & 15);
  const float* src;
  if (m < 3)       src = nW1 + m * 16384 + k * DIM + n;
  else if (m < 6)  src = nW2 + (m - 3) * 16384 + k * DIM + n;
  else if (m < 9)  src = eW1 + (m - 6) * 32896 + k * DIM + n;
  else if (m < 12) src = eW1 + (m - 9) * 32896 + (128 + k) * DIM + n;
  else             src = eW2 + (m - 12) * 16384 + k * DIM + n;
  w16[t] = (_Float16)(*src);
}

__global__ void f32_to_f16_kernel(const float* __restrict__ x,
                                  _Float16* __restrict__ y, int n) {
  int t = blockIdx.x * 256 + threadIdx.x;
  if (t < n) y[t] = (_Float16)x[t];
}

__global__ void zero_kernel(float* __restrict__ p, int n) {
  int t = blockIdx.x * 256 + threadIdx.x;
  if (t < n) p[t] = 0.0f;
}

// ---------------------------------------------------------------------------
// Node MLP: h = relu(x@W1+b1)@W2+b2, computed as h^T = W^T x^T.
// One wave = 16 nodes x 128 features.
// ---------------------------------------------------------------------------
__global__ void __launch_bounds__(256) node_mlp_kernel(
    const _Float16* __restrict__ x16,
    const _Float16* __restrict__ w1g, const _Float16* __restrict__ w2g,
    const float* __restrict__ b1, const float* __restrict__ b2,
    float* __restrict__ h32, _Float16* __restrict__ h16, int nTiles) {
  extern __shared__ char smem[];
  _Float16* sW1 = (_Float16*)smem;            // 16384 halfs
  _Float16* sW2 = sW1 + 16384;                // 16384 halfs
  _Float16* sT  = sW2 + 16384;                // 8 waves * 16*136 halfs
  {
    uint4* d = (uint4*)sW1; const uint4* s = (const uint4*)w1g;
    for (int i = threadIdx.x; i < 2048; i += 256) d[i] = s[i];
    d = (uint4*)sW2; s = (const uint4*)w2g;
    for (int i = threadIdx.x; i < 2048; i += 256) d[i] = s[i];
  }
  __syncthreads();
  const int wave = threadIdx.x >> 5, lane = threadIdx.x & 31;
  const int half = lane >> 4, r = lane & 15;
  const int tile = blockIdx.x * 8 + wave;
  const bool active = tile < nTiles;
  _Float16* myT = sT + wave * (16 * 136);
  v16h bfr[4];
  if (active) {
    const int node = tile * 16 + r;         // lane's column of the D tile
    const _Float16* rp = x16 + (size_t)node * DIM;
#pragma unroll
    for (int kt = 0; kt < 4; ++kt) bfr[kt] = load_b_row(rp, kt, half);
    for (int ft = 0; ft < 8; ++ft) {
      v8f c = load_bias8(b1 + ft * 16 + 8 * half);
#pragma unroll
      for (int kt = 0; kt < 4; ++kt)
        c = wmma16(load_w_frag(sW1, kt, ft, lane), bfr[kt], c);
      *reinterpret_cast<h8*>(myT + r * 136 + ft * 16 + 8 * half) = relu_pack(c);
    }
  }
  __syncthreads();
  if (active) {
    const int node = tile * 16 + r;
    const _Float16* rp = myT + r * 136;
#pragma unroll
    for (int kt = 0; kt < 4; ++kt) bfr[kt] = load_b_row(rp, kt, half);
    float* o32 = h32 + (size_t)node * DIM;
    _Float16* o16 = h16 + (size_t)node * DIM;
    for (int ft = 0; ft < 8; ++ft) {
      v8f c = load_bias8(b2 + ft * 16 + 8 * half);
#pragma unroll
      for (int kt = 0; kt < 4; ++kt)
        c = wmma16(load_w_frag(sW2, kt, ft, lane), bfr[kt], c);
      const int fo = ft * 16 + 8 * half;
      *reinterpret_cast<float4*>(o32 + fo)     = make_float4(c[0], c[1], c[2], c[3]);
      *reinterpret_cast<float4*>(o32 + fo + 4) = make_float4(c[4], c[5], c[6], c[7]);
      h8 v;
#pragma unroll
      for (int i = 0; i < 8; ++i) v[i] = (_Float16)c[i];
      *reinterpret_cast<h8*>(o16 + fo) = v;
    }
  }
}

// ---------------------------------------------------------------------------
// Edge MLP: m = relu([h[dst],h[src],ea]@eW1+eb1)@eW2+eb2, scatter-add to dst.
// One wave = 16 edges; lane's column = its own edge (one dst/src/ea each).
// feat@eW1 = h[dst]@W1a + h[src]@W1b + ea * w_last.
// ---------------------------------------------------------------------------
__global__ void __launch_bounds__(256) edge_mlp_kernel(
    const _Float16* __restrict__ h16,
    const _Float16* __restrict__ w1ag, const _Float16* __restrict__ w1bg,
    const _Float16* __restrict__ w2g,
    const float* __restrict__ eb1, const float* __restrict__ eb2,
    const float* __restrict__ ea, const float* __restrict__ wlast,
    const int* __restrict__ srcIdx, const int* __restrict__ dstIdx,
    float* __restrict__ aggr, int nTiles) {
  extern __shared__ char smem[];
  _Float16* sW1a = (_Float16*)smem;
  _Float16* sW1b = sW1a + 16384;
  _Float16* sW2  = sW1b + 16384;
  _Float16* sT   = sW2 + 16384;
  {
    uint4* d = (uint4*)sW1a; const uint4* s = (const uint4*)w1ag;
    for (int i = threadIdx.x; i < 2048; i += 256) d[i] = s[i];
    d = (uint4*)sW1b; s = (const uint4*)w1bg;
    for (int i = threadIdx.x; i < 2048; i += 256) d[i] = s[i];
    d = (uint4*)sW2; s = (const uint4*)w2g;
    for (int i = threadIdx.x; i < 2048; i += 256) d[i] = s[i];
  }
  __syncthreads();
  const int wave = threadIdx.x >> 5, lane = threadIdx.x & 31;
  const int half = lane >> 4, r = lane & 15;
  const int tile = blockIdx.x * 8 + wave;
  const bool active = tile < nTiles;
  _Float16* myT = sT + wave * (16 * 136);
  v16h bd[4], bs[4];
  int drow = 0;
  float eaval = 0.0f;
  if (active) {
    const int e = tile * 16 + r;
    drow = dstIdx[e];
    const int srow = srcIdx[e];
    eaval = ea[e];
    const _Float16* dp = h16 + (size_t)drow * DIM;
    const _Float16* sp = h16 + (size_t)srow * DIM;
#pragma unroll
    for (int kt = 0; kt < 4; ++kt) {
      bd[kt] = load_b_row(dp, kt, half);
      bs[kt] = load_b_row(sp, kt, half);
    }
    for (int ft = 0; ft < 8; ++ft) {
      const int fo = ft * 16 + 8 * half;
      v8f c = load_bias8(eb1 + fo);
      const v8f wl = load_bias8(wlast + fo);
#pragma unroll
      for (int i = 0; i < 8; ++i) c[i] += eaval * wl[i];
#pragma unroll
      for (int kt = 0; kt < 4; ++kt) {
        c = wmma16(load_w_frag(sW1a, kt, ft, lane), bd[kt], c);
        c = wmma16(load_w_frag(sW1b, kt, ft, lane), bs[kt], c);
      }
      *reinterpret_cast<h8*>(myT + r * 136 + fo) = relu_pack(c);
    }
  }
  __syncthreads();
  if (active) {
    const _Float16* rp = myT + r * 136;
#pragma unroll
    for (int kt = 0; kt < 4; ++kt) bd[kt] = load_b_row(rp, kt, half);
    float* obase = aggr + (size_t)drow * DIM;
    for (int ft = 0; ft < 8; ++ft) {
      const int fo = ft * 16 + 8 * half;
      v8f c = load_bias8(eb2 + fo);
#pragma unroll
      for (int kt = 0; kt < 4; ++kt)
        c = wmma16(load_w_frag(sW2, kt, ft, lane), bd[kt], c);
#pragma unroll
      for (int i = 0; i < 8; ++i) atomic_add_f32(obase + fo + i, c[i]);
    }
  }
}

// ---------------------------------------------------------------------------
// BatchNorm stats: per-column sum / sumsq over all rows.
// ---------------------------------------------------------------------------
__global__ void bn_stats_kernel(const float* __restrict__ x,
                                float* __restrict__ stats,
                                int nRows, int rowsPerBlock) {
  const int col = threadIdx.x & 127;
  const int sub = threadIdx.x >> 7;  // 0 or 1
  const int r0 = blockIdx.x * rowsPerBlock;
  const int r1 = min(nRows, r0 + rowsPerBlock);
  float s = 0.0f, s2 = 0.0f;
  for (int r = r0 + sub; r < r1; r += 2) {
    const float v = x[(size_t)r * DIM + col];
    s += v; s2 += v * v;
  }
  atomic_add_f32(&stats[col], s);
  atomic_add_f32(&stats[128 + col], s2);
}

__global__ void bn_apply_kernel(const float* __restrict__ h,
                                const float* __restrict__ stats,
                                const float* __restrict__ g,
                                const float* __restrict__ b,
                                const float* __restrict__ res,
                                float* __restrict__ y32,
                                _Float16* __restrict__ y16, int hasRes) {
  const int t = blockIdx.x * 256 + threadIdx.x;
  if (t >= N_NODES * DIM) return;
  const int col = t & 127;
  const float invN = 1.0f / (float)N_NODES;
  const float mu = stats[col] * invN;
  const float var = stats[128 + col] * invN - mu * mu;
  const float scale = g[col] * rsqrtf(var + 1e-5f);
  const float shift = b[col] - mu * scale;
  float v = fmaxf(h[t] * scale + shift, 0.0f);
  if (hasRes) v += res[t];
  y32[t] = v;
  y16[t] = (_Float16)v;
}

// ---------------------------------------------------------------------------
// Global mean pool over sorted batch ids (run-length local accumulation).
// ---------------------------------------------------------------------------
__global__ void pool_kernel(const float* __restrict__ x,
                            const int* __restrict__ batch,
                            float* __restrict__ sums, float* __restrict__ cnts,
                            int rowsPerBlock) {
  const int col = threadIdx.x;  // 128 threads
  const int r0 = blockIdx.x * rowsPerBlock;
  const int r1 = min(N_NODES, r0 + rowsPerBlock);
  if (r0 >= r1) return;
  float acc = 0.0f, cnt = 0.0f;
  int curg = batch[r0];
  for (int r = r0; r < r1; ++r) {
    const int gi = batch[r];
    if (gi != curg) {
      atomic_add_f32(&sums[curg * DIM + col], acc);
      if (col == 0) atomic_add_f32(&cnts[curg], cnt);
      acc = 0.0f; cnt = 0.0f; curg = gi;
    }
    acc += x[(size_t)r * DIM + col];
    cnt += 1.0f;
  }
  atomic_add_f32(&sums[curg * DIM + col], acc);
  if (col == 0) atomic_add_f32(&cnts[curg], cnt);
}

__global__ void fc_kernel(const float* __restrict__ sums,
                          const float* __restrict__ cnts,
                          const float* __restrict__ fc1W,
                          const float* __restrict__ fc1b,
                          const float* __restrict__ fc2W,
                          const float* __restrict__ fc2b,
                          float* __restrict__ out) {
  __shared__ float pooled[N_GRAPHS][DIM];
  __shared__ float t1[N_GRAPHS][DIM];
  const int c = threadIdx.x;  // 128
  for (int g = 0; g < N_GRAPHS; ++g)
    pooled[g][c] = sums[g * DIM + c] / fmaxf(cnts[g], 1.0f);
  __syncthreads();
  for (int g = 0; g < N_GRAPHS; ++g) {
    float acc = fc1b[c];
    for (int k = 0; k < DIM; ++k) acc += pooled[g][k] * fc1W[k * DIM + c];
    t1[g][c] = fmaxf(acc, 0.0f);
  }
  __syncthreads();
  if (c < N_GRAPHS) {
    float o = fc2b[0];
    for (int k = 0; k < DIM; ++k) o += t1[c][k] * fc2W[k];
    out[c] = o;
  }
}

// ---------------------------------------------------------------------------
extern "C" void kernel_launch(void* const* d_in, const int* in_sizes, int n_in,
                              void* d_out, int out_size, void* d_ws, size_t ws_size,
                              hipStream_t stream) {
  (void)in_sizes; (void)n_in; (void)out_size; (void)ws_size;
  const float* x    = (const float*)d_in[0];
  const float* ea   = (const float*)d_in[1];
  const float* nW1  = (const float*)d_in[2];
  const float* nb1  = (const float*)d_in[3];
  const float* nW2  = (const float*)d_in[4];
  const float* nb2  = (const float*)d_in[5];
  const float* eW1  = (const float*)d_in[6];
  const float* eb1  = (const float*)d_in[7];
  const float* eW2  = (const float*)d_in[8];
  const float* eb2  = (const float*)d_in[9];
  const float* bng  = (const float*)d_in[10];
  const float* bnb  = (const float*)d_in[11];
  const float* fc1W = (const float*)d_in[12];
  const float* fc1b = (const float*)d_in[13];
  const float* fc2W = (const float*)d_in[14];
  const float* fc2b = (const float*)d_in[15];
  const int* eidx   = (const int*)d_in[16];
  const int* batch  = (const int*)d_in[17];
  const int* srcIdx = eidx;
  const int* dstIdx = eidx + N_EDGES;
  float* out = (float*)d_out;

  // workspace carve-up (all offsets 256B aligned)
  char* ws = (char*)d_ws;
  _Float16* w16 = (_Float16*)ws;      ws += (size_t)15 * 16384 * sizeof(_Float16);
  _Float16* bufA16 = (_Float16*)ws;   ws += (size_t)N_NODES * DIM * sizeof(_Float16);
  _Float16* bufB16 = (_Float16*)ws;   ws += (size_t)N_NODES * DIM * sizeof(_Float16);
  float* h32  = (float*)ws;           ws += (size_t)N_NODES * DIM * sizeof(float);
  float* y32A = (float*)ws;           ws += (size_t)N_NODES * DIM * sizeof(float);
  float* y32B = (float*)ws;           ws += (size_t)N_NODES * DIM * sizeof(float);
  float* stats = (float*)ws;          ws += 256 * sizeof(float);
  float* psum = (float*)ws;           ws += (size_t)N_GRAPHS * DIM * sizeof(float);
  float* pcnt = (float*)ws;           ws += (size_t)N_GRAPHS * sizeof(float);

  prep_weights_kernel<<<(15 * 16384 + 255) / 256, 256, 0, stream>>>(nW1, nW2, eW1, eW2, w16);
  f32_to_f16_kernel<<<(N_NODES * DIM + 255) / 256, 256, 0, stream>>>(x, bufA16, N_NODES * DIM);

  const size_t nodeShm = (size_t)(2 * 16384 + 8 * 16 * 136) * sizeof(_Float16);
  const size_t edgeShm = (size_t)(3 * 16384 + 8 * 16 * 136) * sizeof(_Float16);
  const int nodeTiles = N_NODES / 16;   // 3125
  const int edgeTiles = N_EDGES / 16;   // 50000
  const int nodeBlocks = (nodeTiles + 7) / 8;
  const int edgeBlocks = (edgeTiles + 7) / 8;

  float* targets[3]      = {y32A, y32B, y32A};
  const float* resids[3] = {nullptr, y32A, y32B};

  for (int l = 0; l < 3; ++l) {
    node_mlp_kernel<<<nodeBlocks, 256, nodeShm, stream>>>(
        bufA16, w16 + (size_t)l * 16384, w16 + (size_t)(3 + l) * 16384,
        nb1 + l * DIM, nb2 + l * DIM, h32, bufB16, nodeTiles);
    edge_mlp_kernel<<<edgeBlocks, 256, edgeShm, stream>>>(
        bufB16, w16 + (size_t)(6 + l) * 16384, w16 + (size_t)(9 + l) * 16384,
        w16 + (size_t)(12 + l) * 16384, eb1 + l * DIM, eb2 + l * DIM,
        ea, eW1 + (size_t)l * 32896 + 32768, srcIdx, dstIdx, h32, edgeTiles);
    zero_kernel<<<1, 256, 0, stream>>>(stats, 256);
    bn_stats_kernel<<<250, 256, 0, stream>>>(h32, stats, N_NODES, 200);
    bn_apply_kernel<<<(N_NODES * DIM + 255) / 256, 256, 0, stream>>>(
        h32, stats, bng + l * DIM, bnb + l * DIM, resids[l], targets[l], bufA16,
        (l == 0) ? 0 : 1);
  }

  zero_kernel<<<(N_GRAPHS * DIM + N_GRAPHS + 255) / 256, 256, 0, stream>>>(
      psum, N_GRAPHS * DIM + N_GRAPHS);  // psum and pcnt are contiguous
  pool_kernel<<<250, 128, 0, stream>>>(y32A, batch, psum, pcnt, 200);
  fc_kernel<<<1, 128, 0, stream>>>(psum, pcnt, fc1W, fc1b, fc2W, fc2b, out);
}
